// FullFreqAttention_43465069035681
// MI455X (gfx1250) — matrix operand order repaired and checked by reading, not compiled, not run on hardware
//
#include <hip/hip_runtime.h>

// CDNA5 / gfx1250 flash-attention-style complex attention.
// B=8, L=S=1024, H=8, E=64.  out[b,l,h,d,{re,im}] (fp32).
// bf16 WMMA (v_wmma_f32_16x16x32_bf16), fp32 accumulation, base-2 online softmax,
// double-buffered LDS staging, CNeg for the complex cross term, row-sums via
// P x ones WMMA.

typedef __attribute__((ext_vector_type(16))) __bf16 bf16x16;
typedef __attribute__((ext_vector_type(8)))  float  f32x8;

#define B_DIM 8
#define L_DIM 1024
#define H_DIM 8
#define E_DIM 64
#define STILE 32            // S columns processed per iteration
#define NTILES (L_DIM / STILE)
#define WAVES 8
#define BLOCK_ROWS 128      // L rows per block (8 waves x 16 rows)
// softmax in base 2: SCALE * log2(e) = 0.125 * 1.44269504
#define SCALE_LOG2E 0.1803368801111244f

union FragB16 {
    bf16x16 v;
    uint4 q[2];
    unsigned int u[8];
    unsigned short s[16];
};

__device__ __forceinline__ unsigned short f2bf(float f) {
    __bf16 h = (__bf16)f;                       // native v_cvt on gfx1250
    return __builtin_bit_cast(unsigned short, h);
}
__device__ __forceinline__ unsigned int pack2bf(float lo, float hi) {
    return (unsigned int)f2bf(lo) | ((unsigned int)f2bf(hi) << 16);
}

__device__ __forceinline__ f32x8 wmma_bf16(bf16x16 a, bf16x16 b, f32x8 c) {
    // (neg_a, A, neg_b, B, c_mod, C, reuse_a, reuse_b)
    return __builtin_amdgcn_wmma_f32_16x16x32_bf16(false, a, false, b,
                                                   (short)0, c, false, false);
}
__device__ __forceinline__ f32x8 wmma_bf16_negc(bf16x16 a, bf16x16 b, f32x8 c) {
    // c_mod = 1 -> NEG[2] (negate C-matrix): D = A*B - C
    return __builtin_amdgcn_wmma_f32_16x16x32_bf16(false, a, false, b,
                                                   (short)1, c, false, false);
}

__global__ __launch_bounds__(256)
void ffa_kernel(const float* __restrict__ q_re, const float* __restrict__ q_im,
                const float* __restrict__ k_re, const float* __restrict__ k_im,
                const float* __restrict__ v_re, const float* __restrict__ v_im,
                float* __restrict__ out)
{
    // Double-buffered bf16 staging: K row-major [s][e], V transposed [d][s].
    __shared__ __align__(16) unsigned short ldsKr[2][STILE * E_DIM];
    __shared__ __align__(16) unsigned short ldsKi[2][STILE * E_DIM];
    __shared__ __align__(16) unsigned short ldsVr[2][E_DIM * STILE];
    __shared__ __align__(16) unsigned short ldsVi[2][E_DIM * STILE];
    __shared__ __align__(16) unsigned short ldsP[WAVES * 16 * STILE];

    const int tid  = threadIdx.x;
    const int wave = tid >> 5;
    const int lane = tid & 31;
    const int half = lane >> 4;     // 0: lanes 0-15, 1: lanes 16-31
    const int l16  = lane & 15;

    const int ltile = blockIdx.x;   // 0..7  (128 rows each)
    const int h     = blockIdx.y;   // 0..7
    const int b     = blockIdx.z;   // 0..7

    const int l0 = ltile * BLOCK_ROWS + wave * 16;   // first Q row of this wave

    const int rowStride = H_DIM * E_DIM;             // 512 floats between l steps
    const long base = ((long)b * L_DIM) * rowStride + (long)h * E_DIM;

    // staging role of this thread: one (s_loc, 8-wide e chunk) per matrix
    const int s_loc = tid >> 3;        // 0..31
    const int c8    = (tid & 7) * 8;   // 0,8,..,56

    // constant all-ones bf16 B-fragment (for row sums of P)
    FragB16 ONES;
    #pragma unroll
    for (int i = 0; i < 8; ++i) ONES.u[i] = 0x3F803F80u;   // bf16 1.0 x2

    // ---- Load this wave's Q stripe (16 rows x 64) as bf16 A-fragments ----
    FragB16 Qr[2], Qi[2];
    {
        const int qrow = l0 + l16;
        const float* pr = q_re + base + (long)qrow * rowStride;
        const float* pi = q_im + base + (long)qrow * rowStride;
        #pragma unroll
        for (int f = 0; f < 2; ++f) {
            const int e0 = f * 32 + half * 16;
            const int e1 = f * 32 + half * 16 + 8;
            float4 a0 = *(const float4*)(pr + e0);
            float4 a1 = *(const float4*)(pr + e0 + 4);
            float4 a2 = *(const float4*)(pr + e1);
            float4 a3 = *(const float4*)(pr + e1 + 4);
            Qr[f].u[0] = pack2bf(a0.x, a0.y); Qr[f].u[1] = pack2bf(a0.z, a0.w);
            Qr[f].u[2] = pack2bf(a1.x, a1.y); Qr[f].u[3] = pack2bf(a1.z, a1.w);
            Qr[f].u[4] = pack2bf(a2.x, a2.y); Qr[f].u[5] = pack2bf(a2.z, a2.w);
            Qr[f].u[6] = pack2bf(a3.x, a3.y); Qr[f].u[7] = pack2bf(a3.z, a3.w);
            float4 b0 = *(const float4*)(pi + e0);
            float4 b1 = *(const float4*)(pi + e0 + 4);
            float4 b2 = *(const float4*)(pi + e1);
            float4 b3 = *(const float4*)(pi + e1 + 4);
            Qi[f].u[0] = pack2bf(b0.x, b0.y); Qi[f].u[1] = pack2bf(b0.z, b0.w);
            Qi[f].u[2] = pack2bf(b1.x, b1.y); Qi[f].u[3] = pack2bf(b1.z, b1.w);
            Qi[f].u[4] = pack2bf(b2.x, b2.y); Qi[f].u[5] = pack2bf(b2.z, b2.w);
            Qi[f].u[6] = pack2bf(b3.x, b3.y); Qi[f].u[7] = pack2bf(b3.z, b3.w);
        }
    }

    // ---- Flash-attention state ----
    f32x8 Or[4], Oi[4];
    #pragma unroll
    for (int t = 0; t < 4; ++t) {
        #pragma unroll
        for (int r = 0; r < 8; ++r) { Or[t][r] = 0.f; Oi[t][r] = 0.f; }
    }
    float mrow[8], lrow[8];
    #pragma unroll
    for (int r = 0; r < 8; ++r) { mrow[r] = -1e30f; lrow[r] = 0.f; }

    // ---- Prologue: stage tile 0 into buffer 0 ----
    {
        const long g = base + (long)s_loc * rowStride + c8;
        float4 x0 = *(const float4*)(k_re + g), x1 = *(const float4*)(k_re + g + 4);
        float4 y0 = *(const float4*)(k_im + g), y1 = *(const float4*)(k_im + g + 4);
        float4 u0 = *(const float4*)(v_re + g), u1 = *(const float4*)(v_re + g + 4);
        float4 w0 = *(const float4*)(v_im + g), w1 = *(const float4*)(v_im + g + 4);
        uint4 pk  = { pack2bf(x0.x, x0.y), pack2bf(x0.z, x0.w),
                      pack2bf(x1.x, x1.y), pack2bf(x1.z, x1.w) };
        *(uint4*)&ldsKr[0][s_loc * E_DIM + c8] = pk;
        uint4 pki = { pack2bf(y0.x, y0.y), pack2bf(y0.z, y0.w),
                      pack2bf(y1.x, y1.y), pack2bf(y1.z, y1.w) };
        *(uint4*)&ldsKi[0][s_loc * E_DIM + c8] = pki;
        float vr8[8] = { u0.x, u0.y, u0.z, u0.w, u1.x, u1.y, u1.z, u1.w };
        float vi8[8] = { w0.x, w0.y, w0.z, w0.w, w1.x, w1.y, w1.z, w1.w };
        #pragma unroll
        for (int i = 0; i < 8; ++i) {
            ldsVr[0][(c8 + i) * STILE + s_loc] = f2bf(vr8[i]);   // transpose [d][s]
            ldsVi[0][(c8 + i) * STILE + s_loc] = f2bf(vi8[i]);
        }
    }
    __syncthreads();

    // ---- Stream over S tiles, double-buffered ----
    for (int it = 0; it < NTILES; ++it) {
        const int cur = it & 1;
        const bool haveNext = (it + 1) < NTILES;

        // issue next tile's global loads early (latency hidden behind WMMAs)
        float4 x0, x1, y0, y1, u0, u1, w0, w1;
        if (haveNext) {
            const long g = base + (long)((it + 1) * STILE + s_loc) * rowStride + c8;
            x0 = *(const float4*)(k_re + g); x1 = *(const float4*)(k_re + g + 4);
            y0 = *(const float4*)(k_im + g); y1 = *(const float4*)(k_im + g + 4);
            u0 = *(const float4*)(v_re + g); u1 = *(const float4*)(v_re + g + 4);
            w0 = *(const float4*)(v_im + g); w1 = *(const float4*)(v_im + g + 4);
        }

        // ---- B-fragments of K (lane col n = lane&15, k run = half*16..+15) ----
        FragB16 Kr[2][2], Ki[2][2];
        #pragma unroll
        for (int j = 0; j < 2; ++j) {
            const int scol = j * 16 + l16;
            #pragma unroll
            for (int f = 0; f < 2; ++f) {
                const int e0 = f * 32 + half * 16;
                Kr[j][f].q[0] = *(const uint4*)&ldsKr[cur][scol * E_DIM + e0];
                Kr[j][f].q[1] = *(const uint4*)&ldsKr[cur][scol * E_DIM + e0 + 8];
                Ki[j][f].q[0] = *(const uint4*)&ldsKi[cur][scol * E_DIM + e0];
                Ki[j][f].q[1] = *(const uint4*)&ldsKi[cur][scol * E_DIM + e0 + 8];
            }
        }

        // ---- scores: re = Qr*Kr - Qi*Ki (via CNeg) ; im = Qr*Ki + Qi*Kr ----
        f32x8 Sre[2], Sim[2];
        #pragma unroll
        for (int j = 0; j < 2; ++j) {
            f32x8 cre = {};
            cre = wmma_bf16(Qi[0].v, Ki[j][0].v, cre);
            cre = wmma_bf16(Qi[1].v, Ki[j][1].v, cre);
            cre = wmma_bf16_negc(Qr[0].v, Kr[j][0].v, cre);  // Qr*Kr - (Qi*Ki)
            cre = wmma_bf16(Qr[1].v, Kr[j][1].v, cre);
            f32x8 cim = {};
            cim = wmma_bf16(Qr[0].v, Ki[j][0].v, cim);
            cim = wmma_bf16(Qr[1].v, Ki[j][1].v, cim);
            cim = wmma_bf16(Qi[0].v, Kr[j][0].v, cim);
            cim = wmma_bf16(Qi[1].v, Kr[j][1].v, cim);
            Sre[j] = cre; Sim[j] = cim;
        }

        // ---- base-2 online softmax on SCALE*log2e*|score| ----
        float a0[8], a1[8], alpha[8];
        #pragma unroll
        for (int r = 0; r < 8; ++r) {
            a0[r] = SCALE_LOG2E *
                __builtin_amdgcn_sqrtf(Sre[0][r] * Sre[0][r] + Sim[0][r] * Sim[0][r]);
            a1[r] = SCALE_LOG2E *
                __builtin_amdgcn_sqrtf(Sre[1][r] * Sre[1][r] + Sim[1][r] * Sim[1][r]);
        }
        #pragma unroll
        for (int r = 0; r < 8; ++r) {
            float tmax = fmaxf(a0[r], a1[r]);
            #pragma unroll
            for (int msk = 8; msk >= 1; msk >>= 1)
                tmax = fmaxf(tmax, __shfl_xor(tmax, msk, 32));
            const float mnew = fmaxf(mrow[r], tmax);
            alpha[r] = __builtin_amdgcn_exp2f(mrow[r] - mnew);
            mrow[r]  = mnew;
        }
        // P = exp2(a - m), bf16, into per-wave LDS strip (C-layout -> [row][s])
        #pragma unroll
        for (int r = 0; r < 8; ++r) {
            const float p0 = __builtin_amdgcn_exp2f(a0[r] - mrow[r]);
            const float p1 = __builtin_amdgcn_exp2f(a1[r] - mrow[r]);
            const int prow = half * 8 + r;
            ldsP[(wave * 16 + prow) * STILE + l16]      = f2bf(p0);
            ldsP[(wave * 16 + prow) * STILE + 16 + l16] = f2bf(p1);
        }
        // O rescale (vectorizable to v_pk_mul_f32)
        #pragma unroll
        for (int t = 0; t < 4; ++t) {
            #pragma unroll
            for (int r = 0; r < 8; ++r) { Or[t][r] *= alpha[r]; Oi[t][r] *= alpha[r]; }
        }

        // ---- reload P as 16x32 A-fragment ----
        FragB16 P;
        {
            const int prow = wave * 16 + l16;
            P.q[0] = *(const uint4*)&ldsP[prow * STILE + half * 16];
            P.q[1] = *(const uint4*)&ldsP[prow * STILE + half * 16 + 8];
        }
        // row sums of P via WMMA with all-ones B (replicated across columns)
        f32x8 rs = {};
        rs = wmma_bf16(P.v, ONES.v, rs);

        // ---- accumulate O += P * V ----
        #pragma unroll
        for (int t = 0; t < 4; ++t) {
            FragB16 Vr, Vi;
            const int d = t * 16 + l16;
            const int k0 = half * 16;
            Vr.q[0] = *(const uint4*)&ldsVr[cur][d * STILE + k0];
            Vr.q[1] = *(const uint4*)&ldsVr[cur][d * STILE + k0 + 8];
            Vi.q[0] = *(const uint4*)&ldsVi[cur][d * STILE + k0];
            Vi.q[1] = *(const uint4*)&ldsVi[cur][d * STILE + k0 + 8];
            Or[t] = wmma_bf16(P.v, Vr.v, Or[t]);
            Oi[t] = wmma_bf16(P.v, Vi.v, Oi[t]);
        }

        // l update with WMMA row sums
        #pragma unroll
        for (int r = 0; r < 8; ++r)
            lrow[r] = lrow[r] * alpha[r] + rs[r];

        // ---- store next tile into the other buffer ----
        if (haveNext) {
            const int nxt = cur ^ 1;
            uint4 pk  = { pack2bf(x0.x, x0.y), pack2bf(x0.z, x0.w),
                          pack2bf(x1.x, x1.y), pack2bf(x1.z, x1.w) };
            *(uint4*)&ldsKr[nxt][s_loc * E_DIM + c8] = pk;
            uint4 pki = { pack2bf(y0.x, y0.y), pack2bf(y0.z, y0.w),
                          pack2bf(y1.x, y1.y), pack2bf(y1.z, y1.w) };
            *(uint4*)&ldsKi[nxt][s_loc * E_DIM + c8] = pki;
            float vr8[8] = { u0.x, u0.y, u0.z, u0.w, u1.x, u1.y, u1.z, u1.w };
            float vi8[8] = { w0.x, w0.y, w0.z, w0.w, w1.x, w1.y, w1.z, w1.w };
            #pragma unroll
            for (int i = 0; i < 8; ++i) {
                ldsVr[nxt][(c8 + i) * STILE + s_loc] = f2bf(vr8[i]);
                ldsVi[nxt][(c8 + i) * STILE + s_loc] = f2bf(vi8[i]);
            }
        }
        __syncthreads();
    }

    // ---- epilogue: O / l, write out[b,l,h,d,{re,im}] as float2 ----
    #pragma unroll
    for (int r = 0; r < 8; ++r) {
        const float inv = __builtin_amdgcn_rcpf(lrow[r]);
        const int row = l0 + half * 8 + r;
        #pragma unroll
        for (int t = 0; t < 4; ++t) {
            const int d = t * 16 + l16;
            const long o = ((((long)b * L_DIM + row) * H_DIM + h) * E_DIM + d) * 2;
            float2 val = { Or[t][r] * inv, Oi[t][r] * inv };
            *(float2*)(out + o) = val;
        }
    }
}

extern "C" void kernel_launch(void* const* d_in, const int* in_sizes, int n_in,
                              void* d_out, int out_size, void* d_ws, size_t ws_size,
                              hipStream_t stream) {
    (void)in_sizes; (void)n_in; (void)out_size; (void)d_ws; (void)ws_size;
    const float* q_re = (const float*)d_in[0];
    const float* q_im = (const float*)d_in[1];
    const float* k_re = (const float*)d_in[2];
    const float* k_im = (const float*)d_in[3];
    const float* v_re = (const float*)d_in[4];
    const float* v_im = (const float*)d_in[5];
    float* out = (float*)d_out;

    dim3 grid(L_DIM / BLOCK_ROWS, H_DIM, B_DIM);   // (8, 8, 8)
    dim3 block(256);
    ffa_kernel<<<grid, block, 0, stream>>>(q_re, q_im, k_re, k_im, v_re, v_im, out);
}
